// Encoder_3796751090357
// MI455X (gfx1250) — matrix operand order repaired and checked by reading, not compiled, not run on hardware
//
#include <hip/hip_runtime.h>

#define IN_F 512
#define OUT_F 64

typedef __attribute__((ext_vector_type(16))) __bf16    v16bf;
typedef __attribute__((ext_vector_type(8)))  float     v8f;
typedef __attribute__((ext_vector_type(8)))  unsigned  v8u;

static __device__ __forceinline__ float sigmoidf(float x) {
  return 1.0f / (1.0f + __expf(-x));
}

// Split two f32 into one packed-bf16-hi word (.x) and one packed-bf16-lo
// word (.y). hi = truncate-to-bf16 (exact top 16 bits); lo = trunc-bf16(x-hi).
// Combined representation error ~2^-17 * |x|  (f32-quality for this GEMM).
// v_perm_b32 packs the two high halves in a single instruction.
static __device__ __forceinline__ uint2 split2(float a, float b) {
  unsigned ua = __float_as_uint(a);
  unsigned ub = __float_as_uint(b);
  unsigned hi = __builtin_amdgcn_perm(ub, ua, 0x07060302u);  // {ub.hi16, ua.hi16}
  float ra = a - __uint_as_float(ua & 0xffff0000u);
  float rb = b - __uint_as_float(ub & 0xffff0000u);
  unsigned lo = __builtin_amdgcn_perm(__float_as_uint(rb), __float_as_uint(ra),
                                      0x07060302u);
  return make_uint2(hi, lo);
}

static __device__ __forceinline__ __bf16 bits2bf(unsigned short h) {
  __bf16 o; __builtin_memcpy(&o, &h, 2); return o;
}

// ---------------------------------------------------------------------------
// K1: pack W1 (f32 [512,64]) into WMMA-B-fragment-ordered bf16 hi/lo arrays.
// B layout (16-bit B 32x16, per sparse-doc): lane L holds N = L&15,
// K = (L>>4)*8 + (idx&7) + (idx>>3)*16, idx = 0..15. Record = (kChunk*4+nt)*32+lane,
// 16 bf16 (32 bytes) contiguous -> GEMM loads one fragment as 2x b128.
// ---------------------------------------------------------------------------
__global__ void k_pack_w1(const float* __restrict__ w1,
                          __bf16* __restrict__ ph, __bf16* __restrict__ pl) {
  int t = blockIdx.x * blockDim.x + threadIdx.x;   // 0 .. 32767
  if (t >= (IN_F / 32) * 4 * 32 * 16) return;
  int idx  = t & 15;          // element within fragment
  int lane = (t >> 4) & 31;
  int rec  = t >> 9;          // kChunk*4 + nt
  int nt   = rec & 3;
  int kcI  = rec >> 2;
  int k = kcI * 32 + (lane >> 4) * 8 + (idx & 7) + (idx >> 3) * 16;
  int n = nt * 16 + (lane & 15);
  float x = w1[(size_t)k * OUT_F + n];
  unsigned ux = __float_as_uint(x);
  unsigned hb = ux & 0xffff0000u;
  float r = x - __uint_as_float(hb);
  unsigned lb = __float_as_uint(r);
  ph[t] = bits2bf((unsigned short)(hb >> 16));
  pl[t] = bits2bf((unsigned short)(lb >> 16));
}

// ---------------------------------------------------------------------------
// K2: XW = feat @ W1 via v_wmma_f32_16x16x32_bf16 with hi/lo split (3 WMMAs).
// One wave computes a 16x64 tile (4 N-tiles). Block = 8 waves = 128 rows.
// grid.y selects feat (y=0) vs feat_a (y=1).
// A layout (16-bit A 16x32): M = lane&15, Kbase = (lane>>4)*8,
//   elements 0..7 -> K = Kbase+0..7 (contiguous), 8..15 -> K = Kbase+16..23.
// ---------------------------------------------------------------------------
__global__ __launch_bounds__(256)
void k_gemm(const float* __restrict__ featA, const float* __restrict__ featB,
            const __bf16* __restrict__ w_hi, const __bf16* __restrict__ w_lo,
            float* __restrict__ outA, float* __restrict__ outB, int nrows) {
  const float* src = (blockIdx.y == 0) ? featA : featB;
  float*       dst = (blockIdx.y == 0) ? outA  : outB;

  const int wave = threadIdx.x >> 5;
  const int lane = threadIdx.x & 31;
  const int m0 = (blockIdx.x * 8 + wave) * 16;
  if (m0 >= nrows) return;                 // wave-uniform: EXEC stays all-1s

  int mrow = m0 + (lane & 15);
  int mclamp = mrow < nrows ? mrow : (nrows - 1);
  const int kbA  = (lane >> 4) * 8;
  const int ncol = lane & 15;

  const v16bf* bh_base = (const v16bf*)w_hi + lane;
  const v16bf* bl_base = (const v16bf*)w_lo + lane;

  v8f acc[4] = {};

  for (int kc = 0; kc < IN_F; kc += 32) {
    const float4* arow =
        (const float4*)(src + (size_t)mclamp * IN_F + kc + kbA);
    float4 x0 = arow[0];   // K = kbA + 0..3
    float4 x1 = arow[1];   // K = kbA + 4..7
    float4 x2 = arow[4];   // K = kbA + 16..19
    float4 x3 = arow[5];   // K = kbA + 20..23

    v8u ah, al;
    uint2 s;
    s = split2(x0.x, x0.y); ah[0] = s.x; al[0] = s.y;
    s = split2(x0.z, x0.w); ah[1] = s.x; al[1] = s.y;
    s = split2(x1.x, x1.y); ah[2] = s.x; al[2] = s.y;
    s = split2(x1.z, x1.w); ah[3] = s.x; al[3] = s.y;
    s = split2(x2.x, x2.y); ah[4] = s.x; al[4] = s.y;
    s = split2(x2.z, x2.w); ah[5] = s.x; al[5] = s.y;
    s = split2(x3.x, x3.y); ah[6] = s.x; al[6] = s.y;
    s = split2(x3.z, x3.w); ah[7] = s.x; al[7] = s.y;
    v16bf a_hi = __builtin_bit_cast(v16bf, ah);
    v16bf a_lo = __builtin_bit_cast(v16bf, al);

    const int recbase = (kc >> 5) * 4 * 32;
#pragma unroll
    for (int nt = 0; nt < 4; ++nt) {
      v16bf b_hi = bh_base[recbase + nt * 32];   // one 32B fragment load
      v16bf b_lo = bl_base[recbase + nt * 32];
      acc[nt] = __builtin_amdgcn_wmma_f32_16x16x32_bf16(
          false, a_hi, false, b_hi, (short)0, acc[nt], false, false);
      acc[nt] = __builtin_amdgcn_wmma_f32_16x16x32_bf16(
          false, a_hi, false, b_lo, (short)0, acc[nt], false, false);
      acc[nt] = __builtin_amdgcn_wmma_f32_16x16x32_bf16(
          false, a_lo, false, b_hi, (short)0, acc[nt], false, false);
    }
  }

  // C/D layout: VGPR r -> M = m0 + r + (lane>>4)*8, N = nt*16 + (lane&15)
  const int rbase = m0 + (lane >> 4) * 8;
#pragma unroll
  for (int r = 0; r < 8; ++r) {
    int m = rbase + r;
    if (m < nrows) {
#pragma unroll
      for (int nt = 0; nt < 4; ++nt)
        dst[(size_t)m * OUT_F + nt * 16 + ncol] = acc[nt][r];
    }
  }
}

// ---------------------------------------------------------------------------
// K3: generic f32 zero fill
// ---------------------------------------------------------------------------
__global__ void k_zero(float* __restrict__ p, long long n) {
  long long i = (long long)blockIdx.x * blockDim.x + threadIdx.x;
  if (i < n) p[i] = 0.0f;
}

// ---------------------------------------------------------------------------
// K4/K6: COO SpMM segment-sum via L2-resident fp32 atomics.
// 32 lanes per edge, 2 features per lane (float2 gather + 2 atomic adds),
// done for two feature matrices at once. Optional rowsum (pass 1 only).
// ---------------------------------------------------------------------------
__global__ __launch_bounds__(256)
void k_spmm2(const int* __restrict__ rows, const int* __restrict__ cols,
             const float* __restrict__ vals,
             const float* __restrict__ x1, float* __restrict__ y1,
             const float* __restrict__ x2, float* __restrict__ y2,
             float* __restrict__ rowsum, long long E) {
  long long t = (long long)blockIdx.x * blockDim.x + threadIdx.x;
  long long e = t >> 5;
  if (e >= E) return;
  const int lane = (int)(t & 31);
  const int r = rows[e];
  const int c = cols[e];
  const float v = vals[e];
  const int f = lane * 2;

  float2 g1 = *(const float2*)(x1 + (size_t)c * OUT_F + f);
  unsafeAtomicAdd(y1 + (size_t)r * OUT_F + f,     v * g1.x);
  unsafeAtomicAdd(y1 + (size_t)r * OUT_F + f + 1, v * g1.y);

  float2 g2 = *(const float2*)(x2 + (size_t)c * OUT_F + f);
  unsafeAtomicAdd(y2 + (size_t)r * OUT_F + f,     v * g2.x);
  unsafeAtomicAdd(y2 + (size_t)r * OUT_F + f + 1, v * g2.y);

  if (rowsum && lane == 0) unsafeAtomicAdd(rowsum + r, v);
}

// ---------------------------------------------------------------------------
// K5: emb = relu(z) (z already accumulated in d_out hidden region),
//     emb_a = relu(z_a); also zero vsum/vsum_a (reusing xw buffers).
// ---------------------------------------------------------------------------
__global__ void k_relu_prep(const float* __restrict__ z,
                            const float* __restrict__ z_a,
                            float* __restrict__ emb_out,
                            float* __restrict__ emb_a,
                            float* __restrict__ vsum,
                            float* __restrict__ vsum_a, long long n) {
  long long i = (long long)blockIdx.x * blockDim.x + threadIdx.x;
  if (i >= n) return;
  emb_out[i] = fmaxf(z[i], 0.0f);
  emb_a[i]   = fmaxf(z_a[i], 0.0f);
  vsum[i] = 0.0f;
  vsum_a[i] = 0.0f;
}

// ---------------------------------------------------------------------------
// K7: readout: g = sigmoid(l2norm(vsum/rowsum)) in place, one wave per node.
// ---------------------------------------------------------------------------
__global__ __launch_bounds__(256)
void k_readout(float* __restrict__ vsum, float* __restrict__ vsum_a,
               const float* __restrict__ rowsum, int N) {
  long long t = (long long)blockIdx.x * blockDim.x + threadIdx.x;
  int n = (int)(t >> 5);
  if (n >= N) return;
  const int lane = (int)(t & 31);
  const float rs = rowsum[n];

#pragma unroll
  for (int which = 0; which < 2; ++which) {
    float* buf = which ? vsum_a : vsum;
    float2 v = *(float2*)(buf + (size_t)n * OUT_F + lane * 2);
    v.x /= rs; v.y /= rs;
    float ss = v.x * v.x + v.y * v.y;
#pragma unroll
    for (int off = 16; off > 0; off >>= 1) ss += __shfl_xor(ss, off, 32);
    float d = fmaxf(sqrtf(ss), 1e-12f);
    v.x = sigmoidf(v.x / d);
    v.y = sigmoidf(v.y / d);
    *(float2*)(buf + (size_t)n * OUT_F + lane * 2) = v;
  }
}

// ---------------------------------------------------------------------------
// K8: bilinear discriminator. One wave per node; disc_w staged in LDS with
// stride-65 padding (conflict-free); g/g_a staged per-node in LDS.
// ret[n]   = [emb.W.g + b,   emb_a.W.g + b]
// ret_a[n] = [emb_a.W.g_a+b, emb.W.g_a + b]
// ---------------------------------------------------------------------------
__global__ __launch_bounds__(256)
void k_disc(const float* __restrict__ emb, const float* __restrict__ emb_a,
            const float* __restrict__ g, const float* __restrict__ g_a,
            const float* __restrict__ disc_w, const float* __restrict__ disc_b,
            float* __restrict__ ret, float* __restrict__ ret_a, int N) {
  __shared__ float sW[64 * 65];
  __shared__ float sg[8][64];
  __shared__ float sga[8][64];

  const int wave = threadIdx.x >> 5;
  const int lane = threadIdx.x & 31;
  const int n = blockIdx.x * 8 + wave;

  for (int i = threadIdx.x; i < 64 * 64; i += 256)
    sW[(i >> 6) * 65 + (i & 63)] = disc_w[i];
  if (n < N) {
    sg[wave][lane * 2]      = g[(size_t)n * OUT_F + lane * 2];
    sg[wave][lane * 2 + 1]  = g[(size_t)n * OUT_F + lane * 2 + 1];
    sga[wave][lane * 2]     = g_a[(size_t)n * OUT_F + lane * 2];
    sga[wave][lane * 2 + 1] = g_a[(size_t)n * OUT_F + lane * 2 + 1];
  }
  __syncthreads();
  if (n >= N) return;

  float p0 = 0.f, p1 = 0.f, p2 = 0.f, p3 = 0.f;
#pragma unroll
  for (int ii = 0; ii < 2; ++ii) {
    const int i = lane + ii * 32;
    const float* wr = &sW[i * 65];
    float rg = 0.f, rga = 0.f;
#pragma unroll 8
    for (int j = 0; j < 64; ++j) {
      float wv = wr[j];
      rg  += wv * sg[wave][j];
      rga += wv * sga[wave][j];
    }
    float ei  = emb[(size_t)n * OUT_F + i];
    float eai = emb_a[(size_t)n * OUT_F + i];
    p0 += ei * rg;    // emb . W . g
    p1 += eai * rg;   // emb_a . W . g
    p2 += eai * rga;  // emb_a . W . g_a
    p3 += ei * rga;   // emb . W . g_a
  }
#pragma unroll
  for (int off = 16; off > 0; off >>= 1) {
    p0 += __shfl_xor(p0, off, 32);
    p1 += __shfl_xor(p1, off, 32);
    p2 += __shfl_xor(p2, off, 32);
    p3 += __shfl_xor(p3, off, 32);
  }
  if (lane == 0) {
    const float b = disc_b[0];
    ret[(size_t)n * 2]     = p0 + b;
    ret[(size_t)n * 2 + 1] = p1 + b;
    ret_a[(size_t)n * 2]     = p2 + b;
    ret_a[(size_t)n * 2 + 1] = p3 + b;
  }
}

// ---------------------------------------------------------------------------
extern "C" void kernel_launch(void* const* d_in, const int* in_sizes, int n_in,
                              void* d_out, int out_size, void* d_ws, size_t ws_size,
                              hipStream_t stream) {
  const float* feat    = (const float*)d_in[0];
  const float* feat_a  = (const float*)d_in[1];
  const float* vals    = (const float*)d_in[2];
  const float* weight1 = (const float*)d_in[3];
  // d_in[4] = weight2: dead code in the reference (h is deleted) -> skipped
  const float* disc_w  = (const float*)d_in[5];
  const float* disc_b  = (const float*)d_in[6];
  const int*   rows    = (const int*)d_in[7];
  const int*   cols    = (const int*)d_in[8];

  const int       N  = in_sizes[0] / IN_F;      // 100000
  const long long E  = in_sizes[2];             // 3200000
  const long long NF = (long long)N * OUT_F;    // 6.4M elements

  // Output layout (flat, return order): hidden[N*64], emb[N*64], ret[N*2], ret_a[N*2]
  float* out      = (float*)d_out;
  float* hidden   = out;                         // z accumulates here directly
  float* emb_out  = out + NF;
  float* ret      = out + 2 * NF;
  float* ret_a    = ret + (size_t)N * 2;

  // Workspace layout (buffers reused across phases)
  char* w = (char*)d_ws;
  const size_t W1B = (size_t)IN_F * OUT_F * sizeof(__bf16);   // 64 KB
  const size_t FB  = (size_t)NF * sizeof(float);              // 25.6 MB
  __bf16* w_hi  = (__bf16*)w;
  __bf16* w_lo  = (__bf16*)(w + W1B);
  float*  R1    = (float*)(w + 2 * W1B);            // xw   -> vsum   -> g
  float*  R2    = (float*)(w + 2 * W1B + FB);       // xw_a -> vsum_a -> g_a
  float*  R3    = (float*)(w + 2 * W1B + 2 * FB);   // z_a
  float*  R4    = (float*)(w + 2 * W1B + 3 * FB);   // emb_a
  float*  rsum  = (float*)(w + 2 * W1B + 4 * FB);   // rowsum [N]

  // K1: pack W1 into fragment-ordered bf16 hi/lo
  k_pack_w1<<<(IN_F * OUT_F + 255) / 256, 256, 0, stream>>>(weight1, w_hi, w_lo);

  // K2: xw = feat @ W1, xw_a = feat_a @ W1  (WMMA bf16 split, 3 products)
  dim3 ggrid((N + 127) / 128, 2, 1);
  k_gemm<<<ggrid, 256, 0, stream>>>(feat, feat_a, w_hi, w_lo, R1, R2, N);

  // K3: zero z (in d_out), z_a, rowsum before atomic accumulation
  k_zero<<<(int)((NF + 255) / 256), 256, 0, stream>>>(hidden, NF);
  k_zero<<<(int)((NF + 255) / 256), 256, 0, stream>>>(R3, NF);
  k_zero<<<(N + 255) / 256, 256, 0, stream>>>(rsum, N);

  // K4: z = A@xw (into d_out hidden), z_a = A@xw_a, rowsum = A@1
  long long spmm_threads = E * 32;
  int spmm_blocks = (int)((spmm_threads + 255) / 256);
  k_spmm2<<<spmm_blocks, 256, 0, stream>>>(rows, cols, vals, R1, hidden,
                                           R2, R3, rsum, E);

  // K5: emb = relu(z) -> d_out, emb_a = relu(z_a) -> R4; zero vsum buffers
  k_relu_prep<<<(int)((NF + 255) / 256), 256, 0, stream>>>(
      hidden, R3, emb_out, R4, R1, R2, NF);

  // K6: vsum = A@emb, vsum_a = A@emb_a
  k_spmm2<<<spmm_blocks, 256, 0, stream>>>(rows, cols, vals, emb_out, R1,
                                           R4, R2, nullptr, E);

  // K7: g = sigmoid(l2norm(vsum/rowsum)) in place (R1, R2)
  k_readout<<<(int)(((long long)N * 32 + 255) / 256), 256, 0, stream>>>(
      R1, R2, rsum, N);

  // K8: bilinear discriminator -> ret, ret_a
  k_disc<<<(N + 7) / 8, 256, 0, stream>>>(emb_out, R4, R1, R2, disc_w, disc_b,
                                          ret, ret_a, N);
}
